// ECGMotifGraphModel_18081812316624
// MI455X (gfx1250) — matrix operand order repaired and checked by reading, not compiled, not run on hardware
//
#include <hip/hip_runtime.h>
#include <math.h>

// ---------------------------------------------------------------------------
// Model dims
// ---------------------------------------------------------------------------
#define B_   8
#define N_   1024
#define M_   (B_ * N_)      // 8192 rows
#define DIN_ 128
#define D_   256
#define V_   512
#define H_   8
#define HD_  32
#define GH_  4
#define GHD_ 64
#define E_   16384
#define FF_  1024

typedef __attribute__((ext_vector_type(16))) __bf16 v16bf;
typedef __attribute__((ext_vector_type(4)))  __bf16 v4bf;
typedef __attribute__((ext_vector_type(8)))  float  v8f;

__device__ __forceinline__ float gelu_f(float x) {
  return 0.5f * x * (1.0f + erff(x * 0.70710678118654752f));
}
__device__ __forceinline__ float warp_sum(float v) {
#pragma unroll
  for (int off = 16; off > 0; off >>= 1) v += __shfl_xor(v, off, 32);
  return v;
}
// A-fragment K index map for 16x32 bf16 A (ISA 7.12.2 table)
__device__ __forceinline__ int kmapA(int e, int hi) {
  return (e & 7) + ((e >> 3) << 4) + (hi << 3);
}

// ---------------------------------------------------------------------------
// Generic WMMA GEMM:  C[M,N] = act(A[M,K] @ W[K,N] + bias) (+ residual)
// 64x64 tile per block of 128 threads (4 waves); bf16 staging, f32 accum.
// LDS tiles stored FRAGMENT-MAJOR: slot = (frag*32 + lane)*16 + e, so each
// lane's WMMA fragment is one contiguous 32-byte vector load (ds_load_b128 x2).
// B staging uses 4k x 4n micro-tiles per thread: pairs of consecutive k pack
// into v_cvt_pk_bf16_f32 and land as a single 8-byte ds_store_b64 per column.
// act: 0 = none, 1 = exact GELU
// ---------------------------------------------------------------------------
__global__ void __launch_bounds__(128)
gemm_wmma(const float* __restrict__ A, const float* __restrict__ W,
          const float* __restrict__ bias, const float* __restrict__ residual,
          float* __restrict__ C, int M, int N, int K, int act)
{
  __shared__ alignas(32) __bf16 AsS[4 * 32 * 16];   // 4 row-frags  x 32 lanes x 16
  __shared__ alignas(32) __bf16 BsS[4 * 32 * 16];   // 4 col-frags  x 32 lanes x 16

  const int t  = threadIdx.x;
  const int wv = t >> 5;
  const int lane = t & 31;
  const int lr = lane & 15;
  const int hi = lane >> 4;
  const int m0 = blockIdx.y * 64;
  const int n0 = blockIdx.x * 64;

  // A staging: thread t fills exactly the fragment slot that lane (t&31) of
  // wave (t>>5) consumes: two contiguous 8-float global chunks -> v16bf store.
  const int arow = m0 + (t >> 5) * 16 + lr;
  // B staging: 4k x 4n micro-tile per thread.
  const int bng = t & 15;        // n-group (4 cols each)
  const int bk0 = (t >> 4) * 4;  // k start (0..28)
  const int bhi = (bk0 >> 3) & 1;
  const int be0 = (bk0 & 7) + ((bk0 >> 4) << 3);      // multiple of 4 -> 8B-aligned slot

  v8f acc[4];
#pragma unroll
  for (int cb = 0; cb < 4; ++cb)
#pragma unroll
    for (int j = 0; j < 8; ++j) acc[cb][j] = 0.0f;

  for (int kk = 0; kk < K; kk += 32) {
    // ---- stage A (vectorized global reads, single vector LDS store) ----
    {
      const float* ap = A + (size_t)arow * K + kk + 8 * hi;
      v16bf av;
#pragma unroll
      for (int j = 0; j < 8; ++j) av[j] = (__bf16)ap[j];
#pragma unroll
      for (int j = 0; j < 8; ++j) av[8 + j] = (__bf16)ap[16 + j];
      *(v16bf*)&AsS[t * 16] = av;
    }
    // ---- stage B (4 coalesced b128 row reads, pk-convert, 4x ds_store_b64) --
    {
      const float* wp = W + (size_t)(kk + bk0) * N + n0 + bng * 4;
      float rrow[4][4];
#pragma unroll
      for (int j = 0; j < 4; ++j)
#pragma unroll
        for (int c = 0; c < 4; ++c) rrow[j][c] = wp[(size_t)j * N + c];
#pragma unroll
      for (int c = 0; c < 4; ++c) {
        int n = bng * 4 + c;
        int slot = ((n >> 4) * 32 + bhi * 16 + (n & 15)) * 16 + be0;
        v4bf pk;
        pk[0] = (__bf16)rrow[0][c];
        pk[1] = (__bf16)rrow[1][c];
        pk[2] = (__bf16)rrow[2][c];
        pk[3] = (__bf16)rrow[3][c];
        *(v4bf*)&BsS[slot] = pk;
      }
    }
    // prefetch next K tile into cache while we compute
    if (kk + 32 < K) {
      __builtin_prefetch(A + (size_t)arow * K + kk + 32 + 8 * hi, 0, 0);
      __builtin_prefetch(W + (size_t)(kk + 32 + bk0) * N + n0 + bng * 4, 0, 0);
    }
    __syncthreads();

    const v16bf a = *(const v16bf*)&AsS[(wv * 32 + lane) * 16];
#pragma unroll
    for (int cb = 0; cb < 4; ++cb) {
      const v16bf b = *(const v16bf*)&BsS[(cb * 32 + lane) * 16];
      acc[cb] = __builtin_amdgcn_wmma_f32_16x16x32_bf16(
          false, a, false, b, (short)0, acc[cb], false, false);
    }
    __syncthreads();
  }

#pragma unroll
  for (int cb = 0; cb < 4; ++cb) {
#pragma unroll
    for (int v = 0; v < 8; ++v) {
      int r = m0 + wv * 16 + v + 8 * hi;
      int c = n0 + cb * 16 + lr;
      float val = acc[cb][v];
      if (bias) val += bias[c];
      if (act == 1) val = gelu_f(val);
      if (residual) val += residual[(size_t)r * N + c];
      C[(size_t)r * N + c] = val;
    }
  }
}

// ---------------------------------------------------------------------------
// LayerNorm (wave per row, D multiple of 32). Optional row mask, optional GELU.
// ---------------------------------------------------------------------------
__global__ void __launch_bounds__(256)
ln_kernel(const float* __restrict__ X, const float* __restrict__ g,
          const float* __restrict__ bta, float* __restrict__ Y,
          int M, int D, const unsigned char* __restrict__ mask, int postGelu)
{
  int wv = threadIdx.x >> 5, lane = threadIdx.x & 31;
  int row = blockIdx.x * (blockDim.x >> 5) + wv;
  if (row >= M) return;
  const float* x = X + (size_t)row * D;
  int per = D / 32;
  float s = 0.f;
  for (int j = 0; j < per; ++j) s += x[lane + j * 32];
  float mu = warp_sum(s) / (float)D;
  float v = 0.f;
  for (int j = 0; j < per; ++j) { float d = x[lane + j * 32] - mu; v += d * d; }
  float rs = rsqrtf(warp_sum(v) / (float)D + 1e-5f);
  float mval = mask ? (mask[row] ? 1.f : 0.f) : 1.f;
  float* y = Y + (size_t)row * D;
  for (int j = 0; j < per; ++j) {
    int d = lane + j * 32;
    float o = (x[d] - mu) * rs * g[d] + bta[d];
    if (postGelu) o = gelu_f(o);
    y[d] = o * mval;
  }
}

// ---------------------------------------------------------------------------
// WMMA attention: one block (4 waves) per (16-query tile, head, batch).
// Scores in 64KB LDS strip (f32), softmax, then PV with WMMA.
// QKV layout: [M, 768] = [q(256)|k(256)|v(256)], head h = cols h*32..h*32+31.
// ---------------------------------------------------------------------------
#define ATTN_SMEM_FLOATS (16 * N_ + 4 * 16 * 32 + 128 + 16 + 16)

__global__ void __launch_bounds__(128)
attn_kernel(const float* __restrict__ QKV, const unsigned char* __restrict__ valid,
            float* __restrict__ O)
{
  extern __shared__ float sm[];
  float* S      = sm;                 // [16][1024] scores / probs
  float* Op     = S + 16 * N_;        // [4][16][32] per-wave PV partials
  float* red    = Op + 2048;          // [128]
  float* rowmax = red + 128;          // [16]
  float* rowsum = rowmax + 16;        // [16]

  const int t = threadIdx.x, wv = t >> 5, lane = t & 31;
  const int lr = lane & 15, hi = lane >> 4;
  const int qt = blockIdx.x, h = blockIdx.y, b = blockIdx.z;
  const size_t base = (size_t)b * N_;
  const int qoff = h * HD_, koff = 256 + h * HD_, voff = 512 + h * HD_;

  // ---- Q fragment (16 queries x 32 dims), identical in all waves ----
  v16bf aq;
#pragma unroll
  for (int e = 0; e < 16; ++e)
    aq[e] = (__bf16)QKV[(base + qt * 16 + lr) * 768 + qoff + kmapA(e, hi)];

  // ---- phase 1: scores = QK^T * scale + bias ----
  for (int kb = wv * 16; kb < N_; kb += 64) {
    v16bf bk;
#pragma unroll
    for (int e = 0; e < 16; ++e)
      bk[e] = (__bf16)QKV[(base + kb + lr) * 768 + koff + kmapA(e, hi)];
    v8f s;
#pragma unroll
    for (int j = 0; j < 8; ++j) s[j] = 0.0f;
    s = __builtin_amdgcn_wmma_f32_16x16x32_bf16(false, aq, false, bk,
                                                (short)0, s, false, false);
    int key = kb + lr;
    float bias = valid[base + key] ? 0.0f : -1e9f;
#pragma unroll
    for (int v = 0; v < 8; ++v)
      S[(v + 8 * hi) * N_ + key] = s[v] * 0.17677669529663687f + bias;
  }
  __syncthreads();

  // ---- phase 2: row softmax (8 threads per row, 128 keys each) ----
  {
    int r = t >> 3, sub = t & 7, j0 = sub * 128;
    float m = -1e30f;
    for (int j = 0; j < 128; ++j) m = fmaxf(m, S[r * N_ + j0 + j]);
    red[t] = m;
    __syncthreads();
    if (sub == 0) {
      float mm = red[r * 8];
      for (int i = 1; i < 8; ++i) mm = fmaxf(mm, red[r * 8 + i]);
      rowmax[r] = mm;
    }
    __syncthreads();
    float rm = rowmax[r], sum = 0.f;
    for (int j = 0; j < 128; ++j) {
      float p = __expf(S[r * N_ + j0 + j] - rm);
      S[r * N_ + j0 + j] = p;
      sum += p;
    }
    red[t] = sum;
    __syncthreads();
    if (sub == 0) {
      float ss = 0.f;
      for (int i = 0; i < 8; ++i) ss += red[r * 8 + i];
      rowsum[r] = ss;
    }
    __syncthreads();
  }

  // ---- phase 3: O = P @ V (each wave covers 256 keys) ----
  v8f o0, o1;
#pragma unroll
  for (int j = 0; j < 8; ++j) { o0[j] = 0.f; o1[j] = 0.f; }
  for (int kb = wv * 256; kb < wv * 256 + 256; kb += 32) {
    v16bf ap;
#pragma unroll
    for (int e = 0; e < 16; ++e)
      ap[e] = (__bf16)S[lr * N_ + kb + kmapA(e, hi)];
    v16bf bv0, bv1;
#pragma unroll
    for (int e = 0; e < 16; ++e) {
      const float* vr = QKV + (base + kb + kmapA(e, hi)) * 768 + voff;
      bv0[e] = (__bf16)vr[lr];
      bv1[e] = (__bf16)vr[16 + lr];
    }
    o0 = __builtin_amdgcn_wmma_f32_16x16x32_bf16(false, ap, false, bv0,
                                                 (short)0, o0, false, false);
    o1 = __builtin_amdgcn_wmma_f32_16x16x32_bf16(false, ap, false, bv1,
                                                 (short)0, o1, false, false);
  }
#pragma unroll
  for (int v = 0; v < 8; ++v) {
    int rr = v + 8 * hi;
    Op[(wv * 16 + rr) * 32 + lr]      = o0[v];
    Op[(wv * 16 + rr) * 32 + 16 + lr] = o1[v];
  }
  __syncthreads();

  // ---- combine wave partials, normalize, write ----
  for (int i = t; i < 512; i += 128) {
    int rr = i >> 5, dd = i & 31;
    float s4 = Op[rr * 32 + dd] + Op[(16 + rr) * 32 + dd] +
               Op[(32 + rr) * 32 + dd] + Op[(48 + rr) * 32 + dd];
    O[(base + qt * 16 + rr) * (size_t)D_ + h * HD_ + dd] = s4 / rowsum[rr];
  }
}

// ---------------------------------------------------------------------------
// GAT kernels
// ---------------------------------------------------------------------------
__global__ void gat_logits(const float* __restrict__ Q, const float* __restrict__ K,
                           const int* __restrict__ edges, float* __restrict__ elog)
{
  int idx = blockIdx.x * blockDim.x + threadIdx.x;
  if (idx >= B_ * E_ * GH_) return;
  int h = idx & 3;
  int e = (idx >> 2) & (E_ - 1);
  int b = idx >> 16;
  int src = edges[(size_t)b * 2 * E_ + e];
  int dst = edges[(size_t)b * 2 * E_ + E_ + e];
  const float* q = Q + ((size_t)(b * N_ + dst)) * D_ + h * GHD_;
  const float* k = K + ((size_t)(b * N_ + src)) * D_ + h * GHD_;
  float s = 0.f;
  for (int d = 0; d < GHD_; ++d) s += q[d] * k[d];
  elog[((size_t)(b * GH_ + h)) * E_ + e] = s * 0.125f;
}

__global__ void __launch_bounds__(256)
gat_softmax(float* __restrict__ elog)   // one block per (b,h); global softmax over E
{
  __shared__ float red[256];
  __shared__ float sv;
  float* x = elog + (size_t)blockIdx.x * E_;
  int t = threadIdx.x;
  float m = -1e30f;
  for (int j = t; j < E_; j += 256) m = fmaxf(m, x[j]);
  red[t] = m; __syncthreads();
  for (int s = 128; s > 0; s >>= 1) { if (t < s) red[t] = fmaxf(red[t], red[t + s]); __syncthreads(); }
  if (t == 0) sv = red[0];
  __syncthreads();
  float mx = sv, sum = 0.f;
  for (int j = t; j < E_; j += 256) { float p = __expf(x[j] - mx); x[j] = p; sum += p; }
  red[t] = sum; __syncthreads();
  for (int s = 128; s > 0; s >>= 1) { if (t < s) red[t] += red[t + s]; __syncthreads(); }
  if (t == 0) sv = red[0];
  __syncthreads();
  float inv = 1.f / sv;
  for (int j = t; j < E_; j += 256) x[j] *= inv;
}

__global__ void gat_scatter(const float* __restrict__ V, const int* __restrict__ edges,
                            const float* __restrict__ w, float* __restrict__ oh)
{
  int idx = blockIdx.x * blockDim.x + threadIdx.x;
  if (idx >= B_ * E_ * GH_) return;
  int h = idx & 3;
  int e = (idx >> 2) & (E_ - 1);
  int b = idx >> 16;
  int src = edges[(size_t)b * 2 * E_ + e];
  int dst = edges[(size_t)b * 2 * E_ + E_ + e];
  float ww = w[((size_t)(b * GH_ + h)) * E_ + e];
  const float* v = V + ((size_t)(b * N_ + src)) * D_ + h * GHD_;
  float* o = oh + ((size_t)(b * N_ + dst)) * D_ + h * GHD_;
  for (int d = 0; d < GHD_; ++d) atomicAdd(&o[d], ww * v[d]);
}

// ---------------------------------------------------------------------------
// Elementwise / small kernels
// ---------------------------------------------------------------------------
__global__ void fill_kernel(float* p, float v, size_t n) {
  size_t i = (size_t)blockIdx.x * blockDim.x + threadIdx.x;
  if (i < n) p[i] = v;
}
__global__ void axpby_kernel(const float* X, const float* Y, float* Z,
                             float a, float b, size_t n) {
  size_t i = (size_t)blockIdx.x * blockDim.x + threadIdx.x;
  if (i < n) Z[i] = a * X[i] + (Y ? b * Y[i] : 0.f);
}
__global__ void maskmul_kernel(const float* X, const unsigned char* vm, float* Y,
                               int M, int D) {
  size_t i = (size_t)blockIdx.x * blockDim.x + threadIdx.x;
  if (i < (size_t)M * D) Y[i] = X[i] * (vm[i / D] ? 1.f : 0.f);
}
__global__ void addpos_kernel(const float* X, const float* pos, float* Y) {
  size_t i = (size_t)blockIdx.x * blockDim.x + threadIdx.x;
  if (i < (size_t)M_ * D_) {
    int d = (int)(i % D_);
    int n = (int)((i / D_) % N_);
    Y[i] = X[i] + pos[(size_t)n * D_ + d];
  }
}
__global__ void embed_kernel(const int* __restrict__ tokens, const float* __restrict__ emb,
                             const float* __restrict__ feat,
                             const unsigned char* __restrict__ nodem, float* __restrict__ out) {
  size_t i = (size_t)blockIdx.x * blockDim.x + threadIdx.x;
  if (i >= (size_t)M_ * D_) return;
  int r = (int)(i / D_), d = (int)(i % D_);
  if (nodem[r]) out[i] = emb[(size_t)V_ * D_ + d];
  else          out[i] = emb[(size_t)tokens[r] * D_ + d] + feat[i];
}
__global__ void __launch_bounds__(256)
l2norm_kernel(const float* __restrict__ X, float* __restrict__ Y, int M, int D) {
  int wv = threadIdx.x >> 5, lane = threadIdx.x & 31;
  int row = blockIdx.x * (blockDim.x >> 5) + wv;
  if (row >= M) return;
  const float* x = X + (size_t)row * D;
  float ss = 0.f;
  for (int j = 0; j < D / 32; ++j) { float v = x[lane + j * 32]; ss += v * v; }
  float inv = 1.f / fmaxf(sqrtf(warp_sum(ss)), 1e-12f);
  float* y = Y + (size_t)row * D;
  for (int j = 0; j < D / 32; ++j) y[lane + j * 32] = x[lane + j * 32] * inv;
}
__global__ void __launch_bounds__(256)
pool_reduce(const float* __restrict__ h, const unsigned char* __restrict__ vm,
            float* __restrict__ out) {   // grid B_, block D_
  int b = blockIdx.x, d = threadIdx.x;
  float s = 0.f, cnt = 0.f;
  for (int n = 0; n < N_; ++n) {
    float m = vm[b * N_ + n] ? 1.f : 0.f;
    s += h[((size_t)(b * N_ + n)) * D_ + d] * m;
    cnt += m;
  }
  out[b * D_ + d] = s / fmaxf(cnt, 1e-8f);
}
__global__ void pool_head(const float* __restrict__ X, const float* __restrict__ W,
                          const float* __restrict__ bias, float* __restrict__ out) {
  int idx = blockIdx.x * blockDim.x + threadIdx.x;     // B_*D_ threads
  if (idx >= B_ * D_) return;
  int b = idx >> 8, c = idx & 255;
  float s = bias[c];
  for (int k = 0; k < D_; ++k) s += X[b * D_ + k] * W[k * D_ + c];
  out[b * D_ + c] = gelu_f(s);
}

// ---------------------------------------------------------------------------
// Host orchestration
// ---------------------------------------------------------------------------
extern "C" void kernel_launch(void* const* d_in, const int* in_sizes, int n_in,
                              void* d_out, int out_size, void* d_ws, size_t ws_size,
                              hipStream_t stream)
{
  (void)in_sizes; (void)n_in; (void)out_size; (void)ws_size;
  auto F = [&](int i) { return (const float*)d_in[i]; };

  const int* tokens          = (const int*)d_in[0];
  const float* features      = (const float*)d_in[1];
  const unsigned char* valid = (const unsigned char*)d_in[2];
  const unsigned char* nodem = (const unsigned char*)d_in[3];
  const int* edges           = (const int*)d_in[4];

  const float* emb = F(5);
  const float* pos = F(6);
  const float *fpW1 = F(7), *fpb1 = F(8), *fplng = F(9), *fplnb = F(10),
              *fpW2 = F(11), *fpb2 = F(12);
  const float *inlng = F(13), *inlnb = F(14);
  // seq layers: 15 + 12*l : ln1g ln1b Wqkv bqkv Wo bo ln2g ln2b W1 b1 W2 b2
  // gat layers: 63 + 10*g : Wq bq Wk bk Wv bv Wo bo lng lnb
  const float *flng = F(93), *flnb = F(94);
  const float *fmW1 = F(95), *fmb1 = F(96), *fmW2 = F(97), *fmb2 = F(98);
  const float *tklng = F(99), *tklnb = F(100), *tkW1 = F(101), *tkb1 = F(102),
              *tkW2 = F(103), *tkb2 = F(104);
  const float *prW1 = F(105), *prb1 = F(106), *prW2 = F(107), *prb2 = F(108);
  const float *plng = F(109), *plnb = F(110), *plW = F(111), *plb = F(112);

  // workspace
  float* wsf  = (float*)d_ws;
  float* bufA = wsf;                                  // 8192*1024
  float* bufB = bufA + (size_t)M_ * 1024;             // 8192*1024
  float* xbuf = bufB + (size_t)M_ * 1024;             // 8192*256
  float* xsb  = xbuf + (size_t)M_ * D_;               // 8192*256
  float* pbuf = xsb + (size_t)M_ * D_;                // pooled temps

  // outputs
  float* out_h    = (float*)d_out;
  float* out_log  = out_h + (size_t)M_ * D_;
  float* out_z    = out_log + (size_t)M_ * V_;
  float* out_pool = out_z + (size_t)M_ * D_;
  float* out_hseq = out_pool + (size_t)B_ * D_;
  float* out_hg   = out_hseq + (size_t)M_ * D_;

  auto GEMM = [&](const float* A, const float* W, const float* bias,
                  const float* res, float* C, int M, int N, int K, int act) {
    dim3 g(N / 64, M / 64);
    gemm_wmma<<<g, 128, 0, stream>>>(A, W, bias, res, C, M, N, K, act);
  };
  auto LN = [&](const float* X, const float* g, const float* b, float* Y,
                int M, int D, const unsigned char* mask, int pg) {
    ln_kernel<<<(M + 7) / 8, 256, 0, stream>>>(X, g, b, Y, M, D, mask, pg);
  };
  auto EW = [&](size_t n) { return dim3((unsigned)((n + 255) / 256)); };

  const size_t MD = (size_t)M_ * D_;

  // ---- feature projection + embedding + input LN ----
  GEMM(features, fpW1, fpb1, nullptr, bufA, M_, D_, DIN_, 0);
  LN(bufA, fplng, fplnb, bufB, M_, D_, nullptr, /*gelu*/1);
  GEMM(bufB, fpW2, fpb2, nullptr, bufA, M_, D_, D_, 0);           // feat
  embed_kernel<<<EW(MD), 256, 0, stream>>>(tokens, emb, bufA, nodem, bufB);
  LN(bufB, inlng, inlnb, xbuf, M_, D_, valid, 0);                 // x
  addpos_kernel<<<EW(MD), 256, 0, stream>>>(xbuf, pos, xsb);      // xs = x + pos
  axpby_kernel<<<EW(MD), 256, 0, stream>>>(xbuf, nullptr, out_hg, 1.f, 0.f, MD); // hg = x

  // ---- transformer stack ----
  const size_t attn_smem = ATTN_SMEM_FLOATS * sizeof(float);
  for (int l = 0; l < 4; ++l) {
    int p = 15 + 12 * l;
    const float *ln1g = F(p+0), *ln1b = F(p+1), *Wqkv = F(p+2), *bqkv = F(p+3),
                *Wo = F(p+4), *bo = F(p+5), *ln2g = F(p+6), *ln2b = F(p+7),
                *W1 = F(p+8), *b1 = F(p+9), *W2 = F(p+10), *b2 = F(p+11);
    LN(xsb, ln1g, ln1b, bufA, M_, D_, nullptr, 0);
    GEMM(bufA, Wqkv, bqkv, nullptr, bufB, M_, 3 * D_, D_, 0);     // qkv
    attn_kernel<<<dim3(N_ / 16, H_, B_), 128, attn_smem, stream>>>(bufB, valid, bufA);
    GEMM(bufA, Wo, bo, xsb, xsb, M_, D_, D_, 0);                  // xs += o@Wo+bo
    LN(xsb, ln2g, ln2b, bufA, M_, D_, nullptr, 0);
    GEMM(bufA, W1, b1, nullptr, bufB, M_, FF_, D_, 1);            // gelu FF1
    GEMM(bufB, W2, b2, xsb, xsb, M_, D_, FF_, 0);                 // xs += FF2
  }
  maskmul_kernel<<<EW(MD), 256, 0, stream>>>(xsb, valid, out_hseq, M_, D_);

  // ---- GAT stack (hg lives in out_hg) ----
  float* Qg = bufA;
  float* Kg = bufA + MD;
  float* Vg = bufA + 2 * MD;
  float* tg = bufA + 3 * MD;
  float* elog = bufB;                       // B*GH*E = 524288 floats
  float* oh   = bufB + (1u << 20);          // 8192*256
  for (int l = 0; l < 3; ++l) {
    int p = 63 + 10 * l;
    const float *Wq = F(p+0), *bq = F(p+1), *Wk = F(p+2), *bk = F(p+3),
                *Wv = F(p+4), *bv = F(p+5), *Wo = F(p+6), *bo = F(p+7),
                *lg = F(p+8), *lb = F(p+9);
    GEMM(out_hg, Wq, bq, nullptr, Qg, M_, D_, D_, 0);
    GEMM(out_hg, Wk, bk, nullptr, Kg, M_, D_, D_, 0);
    GEMM(out_hg, Wv, bv, nullptr, Vg, M_, D_, D_, 0);
    gat_logits<<<(B_ * E_ * GH_) / 256, 256, 0, stream>>>(Qg, Kg, edges, elog);
    gat_softmax<<<B_ * GH_, 256, 0, stream>>>(elog);
    fill_kernel<<<EW(MD), 256, 0, stream>>>(oh, 0.f, MD);
    gat_scatter<<<(B_ * E_ * GH_) / 256, 256, 0, stream>>>(Vg, edges, elog, oh);
    GEMM(oh, Wo, bo, out_hg, tg, M_, D_, D_, 0);                  // hg + oh@Wo+bo
    LN(tg, lg, lb, out_hg, M_, D_, valid, 0);
  }

  // ---- merge + final MLP + heads ----
  axpby_kernel<<<EW(MD), 256, 0, stream>>>(out_hseq, out_hg, xbuf, 0.5f, 0.5f, MD);
  GEMM(xbuf, fmW1, fmb1, nullptr, bufA, M_, D_, D_, 1);
  GEMM(bufA, fmW2, fmb2, xbuf, bufB, M_, D_, D_, 0);
  LN(bufB, flng, flnb, out_h, M_, D_, valid, 0);                  // h

  LN(out_h, tklng, tklnb, bufA, M_, D_, nullptr, 0);
  GEMM(bufA, tkW1, tkb1, nullptr, bufB, M_, D_, D_, 1);
  GEMM(bufB, tkW2, tkb2, nullptr, out_log, M_, V_, D_, 0);        // logits

  GEMM(out_h, prW1, prb1, nullptr, bufA, M_, D_, D_, 1);
  GEMM(bufA, prW2, prb2, nullptr, bufB, M_, D_, D_, 0);
  l2norm_kernel<<<(M_ + 7) / 8, 256, 0, stream>>>(bufB, out_z, M_, D_);

  pool_reduce<<<B_, D_, 0, stream>>>(out_h, valid, pbuf);
  LN(pbuf, plng, plnb, pbuf + B_ * D_, B_, D_, nullptr, 0);
  pool_head<<<(B_ * D_ + 255) / 256, 256, 0, stream>>>(pbuf + B_ * D_, plW, plb, out_pool);
}
